// TemporalEventGNNJODIE_56075093016820
// MI455X (gfx1250) — matrix operand order repaired and checked by reading, not compiled, not run on hardware
//
#include <hip/hip_runtime.h>

#define NU_  500000
#define E_   128
#define F_   172
#define MSG_ 428
#define KPAD 448
#define BB_  65536

typedef __bf16 bf16_t;
typedef __attribute__((ext_vector_type(16))) __bf16 v16bf;
typedef __attribute__((ext_vector_type(8)))  float  v8f;

struct Bytes32 { uint4 lo, hi; };

__device__ __forceinline__ bf16_t f2bf(float f) {
    unsigned u = __builtin_bit_cast(unsigned, f);
    u += 0x7fffu + ((u >> 16) & 1u);                 // round-to-nearest-even
    unsigned short s = (unsigned short)(u >> 16);
    return __builtin_bit_cast(bf16_t, s);
}

__device__ __forceinline__ float sigmoidf_(float x) { return 1.f / (1.f + __expf(-x)); }

// A-matrix tile loader (16x32 bf16) from LDS, per ISA 16-bit A layout:
// lane: M = lane&15, half = lane>>4; elements 0..7 -> K = kbase+8*half+0..7,
// elements 8..15 -> K = kbase+16+8*half+0..7  (two contiguous 16B chunks).
__device__ __forceinline__ v16bf load_A_lds(const bf16_t* base, int stride,
                                            int m, int kbase, int half) {
    const bf16_t* p = base + m * stride + kbase + 8 * half;
    Bytes32 t;
    t.lo = *(const uint4*)p;
    t.hi = *(const uint4*)(p + 16);
    return __builtin_bit_cast(v16bf, t);
}

// B-matrix tile loader (32x16 bf16, B = W^T) from global row-major W[N][ldk]:
// lane: N = n, khalf = lane>>4; elements 0..7 -> K = kbase+8*khalf+0..7,
// elements 8..15 -> K = kbase+16+8*khalf+0..7 (two 16B chunks).
__device__ __forceinline__ v16bf load_B_glb(const bf16_t* w, int ldk,
                                            int n, int kbase, int khalf) {
    const bf16_t* p = w + (size_t)n * ldk + kbase + 8 * khalf;
    Bytes32 t;
    t.lo = *(const uint4*)p;
    t.hi = *(const uint4*)(p + 16);
    return __builtin_bit_cast(v16bf, t);
}

#define WMMA_BF16(acc, a, b) \
    __builtin_amdgcn_wmma_f32_16x16x32_bf16(false, (a), false, (b), (short)0, (acc), false, false)

// ---------------- weight conversion: f32 -> bf16, pad K dim with zeros -------
__global__ void convert_pad_kernel(const float* __restrict__ src, bf16_t* __restrict__ dst,
                                   int rows, int csrc, int cdst) {
    int i = blockIdx.x * blockDim.x + threadIdx.x;
    if (i >= rows * cdst) return;
    int r = i / cdst, c = i % cdst;
    float v = (c < csrc) ? src[(size_t)r * csrc + c] : 0.f;
    dst[i] = f2bf(v);
}

// ---------------- main fused kernel: 16 events per workgroup ----------------
// dynamic LDS layout (bytes):
//   0      xu   [16][448] bf16   (14336)
//   14336  xi   [16][448] bf16   (14336)
//   28672  cat  [2][16][256] bf16 (16384)  cols 0:128 = old(h) bf16, 128:256 = new bf16
//   45056  qy   [16][128] bf16   (4096)
//   49152  oldf [2][16][128] f32 (32768)
//   81920  newf [2][16][128] f32 (32768)
//   114688 giS  [16][384] f32    (24576)   (reused as gate acc [16][128])
//   139264 ghS  [16][384] f32    (24576)
//   total 163840
__global__ __launch_bounds__(256, 1)
void jodie_main_kernel(const int* __restrict__ user_ids, const int* __restrict__ item_ids,
                       const float* __restrict__ ts, const float* __restrict__ features,
                       const int* __restrict__ qtime,
                       const float* __restrict__ memory, const float* __restrict__ last_time,
                       const float* __restrict__ time_w,
                       const bf16_t* __restrict__ u_wih_bf, const bf16_t* __restrict__ u_whh_bf,
                       const float* __restrict__ u_bih, const float* __restrict__ u_bhh,
                       const bf16_t* __restrict__ i_wih_bf, const bf16_t* __restrict__ i_whh_bf,
                       const float* __restrict__ i_bih, const float* __restrict__ i_bhh,
                       const bf16_t* __restrict__ gate_w_bf, const float* __restrict__ gate_b,
                       const bf16_t* __restrict__ pred_w_bf, const float* __restrict__ pred_b,
                       float* __restrict__ out)
{
    extern __shared__ char smem[];
    bf16_t* xu   = (bf16_t*)(smem);
    bf16_t* xi   = (bf16_t*)(smem + 14336);
    bf16_t* cat  = (bf16_t*)(smem + 28672);
    bf16_t* qy   = (bf16_t*)(smem + 45056);
    float*  oldf = (float*)(smem + 49152);
    float*  newf = (float*)(smem + 81920);
    float*  giS  = (float*)(smem + 114688);
    float*  ghS  = (float*)(smem + 139264);
    __shared__ int   s_uid[16], s_iid[16];
    __shared__ float s_du[16], s_di[16], s_dq[16];

    const int tid  = threadIdx.x;
    const int lane = tid & 31;
    const int wv   = tid >> 5;          // wave 0..7
    const int b0   = blockIdx.x * 16;

    if (tid < 16) {
        int uid  = user_ids[b0 + tid];
        int iidn = item_ids[b0 + tid] + NU_;
        float t  = ts[b0 + tid];
        s_uid[tid] = uid;
        s_iid[tid] = iidn;
        float lu = last_time[uid];
        s_du[tid] = t - lu;
        s_di[tid] = t - last_time[iidn];
        s_dq[tid] = (float)qtime[0] - lu;
    }
    __syncthreads();

    // ---- gather embeddings, time projection, build bf16 tiles ----
    for (int j = 0; j < 8; ++j) {
        int e = tid + 256 * j;          // 0..2047 over [r in 0..15][c in 0..127]
        int r = e >> 7, c = e & 127;
        float ou = memory[(size_t)s_uid[r] * E_ + c];
        float oi = memory[(size_t)s_iid[r] * E_ + c];
        float tw = time_w[c];
        float pu = ou * (1.f + s_du[r] * tw);
        float pi = oi * (1.f + s_di[r] * tw);
        float qu = ou * (1.f + s_dq[r] * tw);
        oldf[(0 * 16 + r) * E_ + c] = ou;
        oldf[(1 * 16 + r) * E_ + c] = oi;
        cat[(0 * 16 + r) * 256 + c] = f2bf(ou);   // h for user GRU / gate concat lo
        cat[(1 * 16 + r) * 256 + c] = f2bf(oi);
        qy[r * E_ + c] = f2bf(qu);
        xu[r * KPAD + c]        = f2bf(pu);
        xu[r * KPAD + 128 + c]  = f2bf(pi);
        xi[r * KPAD + c]        = f2bf(pi);
        xi[r * KPAD + 128 + c]  = f2bf(pu);
    }
    // features cols 256..427, zero pad 428..447
    for (int s = tid; s < 16 * 192; s += 256) {
        int r  = s / 192;
        int cc = 256 + (s % 192);
        float v = (cc < 256 + F_) ? features[(size_t)(b0 + r) * F_ + (cc - 256)] : 0.f;
        bf16_t bv = f2bf(v);
        xu[r * KPAD + cc] = bv;
        xi[r * KPAD + cc] = bv;
    }
    __syncthreads();

    const int m_ = lane & 15;
    const int hf = lane >> 4;
    const int nl = lane & 15;
    const int rb = hf * 8;

    // ---- two GRUs (g=0 user, g=1 item) ----
    for (int g = 0; g < 2; ++g) {
        const bf16_t* x    = g ? xi : xu;
        const bf16_t* wih  = g ? i_wih_bf : u_wih_bf;
        const bf16_t* whh  = g ? i_whh_bf : u_whh_bf;
        const float*  bih  = g ? i_bih : u_bih;
        const float*  bhh  = g ? i_bhh : u_bhh;
        const bf16_t* hmat = cat + g * 16 * 256;   // h in cols 0..127, stride 256

        for (int t = 0; t < 3; ++t) {              // 24 N-tiles over 8 waves
            int n0 = 16 * (wv + 8 * t);
            v8f acc = {};
            for (int k = 0; k < KPAD; k += 32) {
                v16bf a = load_A_lds(x, KPAD, m_, k, hf);
                v16bf b = load_B_glb(wih, KPAD, n0 + nl, k, hf);
                acc = WMMA_BF16(acc, a, b);
            }
            #pragma unroll
            for (int v = 0; v < 8; ++v)
                giS[(rb + v) * 384 + n0 + nl] = acc[v];

            v8f acch = {};
            for (int k = 0; k < 128; k += 32) {
                v16bf a = load_A_lds(hmat, 256, m_, k, hf);
                v16bf b = load_B_glb(whh, 128, n0 + nl, k, hf);
                acch = WMMA_BF16(acch, a, b);
            }
            #pragma unroll
            for (int v = 0; v < 8; ++v)
                ghS[(rb + v) * 384 + n0 + nl] = acch[v];
        }
        __syncthreads();

        // elementwise GRU combine (f32)
        for (int j = 0; j < 8; ++j) {
            int e = tid + 256 * j;
            int r = e >> 7, c = e & 127;
            float ir = giS[r * 384 + c]        + bih[c];
            float iz = giS[r * 384 + 128 + c]  + bih[128 + c];
            float in = giS[r * 384 + 256 + c]  + bih[256 + c];
            float hr = ghS[r * 384 + c]        + bhh[c];
            float hz = ghS[r * 384 + 128 + c]  + bhh[128 + c];
            float hn = ghS[r * 384 + 256 + c]  + bhh[256 + c];
            float rg = sigmoidf_(ir + hr);
            float zg = sigmoidf_(iz + hz);
            float ng = tanhf(in + rg * hn);
            float h  = oldf[(g * 16 + r) * E_ + c];
            float nh = (1.f - zg) * ng + zg * h;
            newf[(g * 16 + r) * E_ + c] = nh;
            cat[(g * 16 + r) * 256 + 128 + c] = f2bf(nh);   // concat hi half
        }
        __syncthreads();
    }

    // ---- memory gate: sigmoid([old,new] @ gate_w^T + b), blend, write out ----
    for (int g = 0; g < 2; ++g) {
        const bf16_t* a_base = cat + g * 16 * 256;
        int n0 = 16 * wv;                      // 8 waves cover N=128
        v8f acc = {};
        for (int k = 0; k < 256; k += 32) {
            v16bf a = load_A_lds(a_base, 256, m_, k, hf);
            v16bf b = load_B_glb(gate_w_bf, 256, n0 + nl, k, hf);
            acc = WMMA_BF16(acc, a, b);
        }
        #pragma unroll
        for (int v = 0; v < 8; ++v)
            giS[(rb + v) * E_ + n0 + nl] = acc[v];   // reuse giS as gate acc
        __syncthreads();

        for (int j = 0; j < 8; ++j) {
            int e = tid + 256 * j;
            int r = e >> 7, c = e & 127;
            float gv = sigmoidf_(giS[r * E_ + c] + gate_b[c]);
            float o  = oldf[(g * 16 + r) * E_ + c];
            float nh = newf[(g * 16 + r) * E_ + c];
            out[(size_t)(1 + g) * BB_ * E_ + (size_t)(b0 + r) * E_ + c]
                = gv * nh + (1.f - gv) * o;
        }
        __syncthreads();
    }

    // ---- prediction: query_user @ pred_w^T + pred_b (written straight out) ----
    {
        int n0 = 16 * wv;
        v8f acc = {};
        for (int k = 0; k < 128; k += 32) {
            v16bf a = load_A_lds(qy, 128, m_, k, hf);
            v16bf b = load_B_glb(pred_w_bf, 128, n0 + nl, k, hf);
            acc = WMMA_BF16(acc, a, b);
        }
        float pb = pred_b[n0 + nl];
        #pragma unroll
        for (int v = 0; v < 8; ++v)
            out[(size_t)(b0 + rb + v) * E_ + n0 + nl] = acc[v] + pb;
    }
}

// ---------------- launch ----------------
extern "C" void kernel_launch(void* const* d_in, const int* in_sizes, int n_in,
                              void* d_out, int out_size, void* d_ws, size_t ws_size,
                              hipStream_t stream) {
    const int*   user_ids = (const int*)d_in[0];
    const int*   item_ids = (const int*)d_in[1];
    const float* ts       = (const float*)d_in[2];
    const float* features = (const float*)d_in[3];
    const int*   qtime    = (const int*)d_in[4];
    const float* memory   = (const float*)d_in[5];
    const float* last_t   = (const float*)d_in[6];
    const float* time_w   = (const float*)d_in[7];
    const float* u_wih    = (const float*)d_in[8];
    const float* u_whh    = (const float*)d_in[9];
    const float* u_bih    = (const float*)d_in[10];
    const float* u_bhh    = (const float*)d_in[11];
    const float* i_wih    = (const float*)d_in[12];
    const float* i_whh    = (const float*)d_in[13];
    const float* i_bih    = (const float*)d_in[14];
    const float* i_bhh    = (const float*)d_in[15];
    const float* gate_w   = (const float*)d_in[16];
    const float* gate_b   = (const float*)d_in[17];
    const float* pred_w   = (const float*)d_in[18];
    const float* pred_b   = (const float*)d_in[19];
    float* out = (float*)d_out;

    // bf16 weight staging in workspace (byte offsets, 512B-aligned)
    char* ws = (char*)d_ws;
    bf16_t* u_wih_bf  = (bf16_t*)(ws + 0);        // 384x448
    bf16_t* i_wih_bf  = (bf16_t*)(ws + 344064);   // 384x448
    bf16_t* u_whh_bf  = (bf16_t*)(ws + 688128);   // 384x128
    bf16_t* i_whh_bf  = (bf16_t*)(ws + 786432);   // 384x128
    bf16_t* gate_w_bf = (bf16_t*)(ws + 884736);   // 128x256
    bf16_t* pred_w_bf = (bf16_t*)(ws + 950272);   // 128x128

    auto cvt = [&](const float* s, bf16_t* d, int rows, int cs, int cd) {
        int n = rows * cd;
        convert_pad_kernel<<<(n + 255) / 256, 256, 0, stream>>>(s, d, rows, cs, cd);
    };
    cvt(u_wih,  u_wih_bf,  384, MSG_, KPAD);
    cvt(i_wih,  i_wih_bf,  384, MSG_, KPAD);
    cvt(u_whh,  u_whh_bf,  384, E_,   E_);
    cvt(i_whh,  i_whh_bf,  384, E_,   E_);
    cvt(gate_w, gate_w_bf, E_,  2 * E_, 2 * E_);
    cvt(pred_w, pred_w_bf, E_,  E_,   E_);

    const size_t lds_bytes = 163840;
    (void)hipFuncSetAttribute((const void*)jodie_main_kernel,
                              hipFuncAttributeMaxDynamicSharedMemorySize,
                              (int)lds_bytes);
    jodie_main_kernel<<<BB_ / 16, 256, lds_bytes, stream>>>(
        user_ids, item_ids, ts, features, qtime, memory, last_t, time_w,
        u_wih_bf, u_whh_bf, u_bih, u_bhh,
        i_wih_bf, i_whh_bf, i_bih, i_bhh,
        gate_w_bf, gate_b, pred_w_bf, pred_b, out);
}
